// PEM_3874060501717
// MI455X (gfx1250) — matrix operand order repaired and checked by reading, not compiled, not run on hardware
//
#include <hip/hip_runtime.h>

typedef __attribute__((ext_vector_type(16))) _Float16 v16h;
typedef __attribute__((ext_vector_type(8)))  _Float16 v8h;
typedef __attribute__((ext_vector_type(8)))  float    v8f;
typedef __attribute__((ext_vector_type(4)))  float    v4f;

#define NPATH    16384
#define DIM      128
#define G3       384
#define LMAX     20
#define BP       16
#define NTHREADS 128
#define ROWP     136    // padded row stride (halves): 272B -> bank-conflict-free A loads
#define GROWP    132    // padded row stride (floats) for g buffer

// ---- LDS layout (bytes) ----
#define OFF_WG   0                        // 96 frag * 1024B = 98304
#define OFF_WR   (OFF_WG + 98304)         // 98304
#define OFF_HS   (OFF_WR + 98304)         // 20*16*ROWP*2 = 87040
#define OFF_H    (OFF_HS + 87040)         // 16*ROWP*2 = 4352
#define OFF_X    (OFF_H  + 4352)          // 2*16*ROWP*2 = 8704
#define OFF_G    (OFF_X  + 8704)          // 16*GROWP*4 = 8448
#define OFF_ATT  (OFF_G  + 8448)          // 16*20*4 = 1280
#define OFF_S    (OFF_ATT + 1280)         // 16*ROWP*2 = 4352
#define OFF_EB   (OFF_S  + 4352)          // 64
#define OFF_LEN  (OFF_EB + 64)            // 64
#define OFF_OFF  (OFF_LEN + 64)           // 64
#define LDS_BYTES (OFF_OFF + 64)          // 310976 <= 320KB
// overlays into the Wg region once the GRU is done:
#define OFF_MT   0                        // 32 frag * 1024B = 32768
#define OFF_WV   32768                    // 32768

__device__ __forceinline__ v16h cat16(v8h lo, v8h hi) {
  return __builtin_shufflevector(lo, hi, 0,1,2,3,4,5,6,7,8,9,10,11,12,13,14,15);
}

__device__ __forceinline__ v8f wmma16(v16h a, v16h b, v8f c) {
  return __builtin_amdgcn_wmma_f32_16x16x32_f16(false, a, false, b, (short)0, c,
                                                false, false);
}

__device__ __forceinline__ float fast_sigmoid(float x) {
  return __builtin_amdgcn_rcpf(1.f + __expf(-x));
}
__device__ __forceinline__ float fast_tanh(float x) {
  float ex = __expf(-2.f * x);
  return (1.f - ex) * __builtin_amdgcn_rcpf(1.f + ex);
}

// A operand (16x32 f16) from a row-major [16][ROWP] f16 LDS region at K-offset 32*kk.
// Layout: lane<16 -> M=lane, halves j: K = j; lane>=16: K += 16.
__device__ __forceinline__ v16h load_a(const _Float16* base, int lane, int kk) {
  const _Float16* p = base + (lane & 15) * ROWP + kk * 32 + ((lane >> 4) << 4);
  v8h lo = *(const v8h*)p;
  v8h hi = *(const v8h*)(p + 8);
  return cat16(lo, hi);
}

// B operand (32x16 f16) from pre-swizzled fragment store: frag f, per lane two
// phase-split 16B chunks (halves j=0..7 at L*8, j=8..15 at 256+L*8).
__device__ __forceinline__ v16h load_b(const _Float16* fragbase, int f, int lane) {
  const _Float16* p = fragbase + f * 512 + lane * 8;
  v8h lo = *(const v8h*)p;
  v8h hi = *(const v8h*)(p + 256);
  return cat16(lo, hi);
}

// ---- tiny precompute kernels ----
__global__ __launch_bounds__(256) void pem_scan(const int* __restrict__ lens,
                                                int* __restrict__ offs) {
  __shared__ int part[256];
  int tid = threadIdx.x;
  int s = 0;
  for (int j = 0; j < 64; ++j) s += lens[tid * 64 + j];
  part[tid] = s;
  __syncthreads();
  if (tid == 0) {
    int run = 0;
    for (int i = 0; i < 256; ++i) { int t = part[i]; part[i] = run; run += t; }
  }
  __syncthreads();
  int base = part[tid];
  for (int j = 0; j < 64; ++j) { offs[tid * 64 + j] = base; base += lens[tid * 64 + j]; }
}

__global__ __launch_bounds__(256) void pem_mt(const float* __restrict__ wq,
                                              const float* __restrict__ wk,
                                              float* __restrict__ mt) {
  int idx = blockIdx.x * 256 + threadIdx.x;   // 16384 elements
  int k = idx >> 7, n = idx & 127;
  float acc = 0.f;
  for (int e = 0; e < DIM; ++e) acc += wq[k * DIM + e] * wk[n * DIM + e];
  mt[idx] = acc;                              // Mt[k][n] = sum_e Wq[k,e]*Wk[n,e]
}

// ---- main fused GRU + attention kernel: 16 paths / block, 4 waves ----
__global__ __launch_bounds__(NTHREADS) void pem_main(
    const float* __restrict__ inputs,  // [4096][128]
    const int*   __restrict__ paths,
    const int*   __restrict__ lens,
    const float* __restrict__ gk,      // [128][384]
    const float* __restrict__ rk,      // [128][384]
    const float* __restrict__ gb,      // [2][384]
    const float* __restrict__ wvp,     // [128][128]
    const int*   __restrict__ offs,    // [16384] path start offsets (ws)
    const float* __restrict__ mt,      // [128][128] = Wq@Wk^T (ws)
    float* __restrict__ out) {         // [16384][128]
  extern __shared__ char smem[];
  const int tid  = threadIdx.x;
  const int lane = tid & 31;
  const int wid  = tid >> 5;                 // wave 0..3 owns columns [32w,32w+32)
  const int hi   = (lane >> 4);              // 0/1 lane half-group
  const int ln   = lane & 15;
  const int pbase = blockIdx.x * BP;

  _Float16* wgF = (_Float16*)(smem + OFF_WG);
  _Float16* wrF = (_Float16*)(smem + OFF_WR);
  _Float16* hsL = (_Float16*)(smem + OFF_HS);
  _Float16* hL  = (_Float16*)(smem + OFF_H);
  _Float16* xL  = (_Float16*)(smem + OFF_X);
  float*    gL  = (float*)(smem + OFF_G);
  float*    attL= (float*)(smem + OFF_ATT);
  _Float16* sL  = (_Float16*)(smem + OFF_S);
  int* ebuf = (int*)(smem + OFF_EB);
  int* lenL = (int*)(smem + OFF_LEN);
  int* offL = (int*)(smem + OFF_OFF);

  // ---- stage Wg/Wr as f16 WMMA B fragments (frag f = kk*24+nt) ----
  for (int w = tid; w < 4 * 24 * 32 * 8; w += NTHREADS) {
    int word = w & 7, l2 = (w >> 3) & 31, f = w >> 8;
    int kk = f / 24, nt = f % 24;
    int j0 = word * 2;
    int k0 = kk * 32 + ((l2 >> 4) << 4) + j0;     // K = j + 16*(lane>>4)
    int n  = nt * 16 + (l2 & 15);
    int dst = f * 512 + ((j0 >> 3) ? 256 : 0) + l2 * 8 + (j0 & 7);
    wgF[dst]     = (_Float16)gk[k0 * G3 + n];
    wgF[dst + 1] = (_Float16)gk[(k0 + 1) * G3 + n];
    wrF[dst]     = (_Float16)rk[k0 * G3 + n];
    wrF[dst + 1] = (_Float16)rk[(k0 + 1) * G3 + n];
  }
  if (tid < BP) { offL[tid] = offs[pbase + tid]; lenL[tid] = lens[pbase + tid]; }
  for (int i = tid; i < BP * ROWP; i += NTHREADS) hL[i] = (_Float16)0.f;
  __syncthreads();
  if (tid < BP) ebuf[tid] = (0 < lenL[tid]) ? paths[offL[tid]] : -1;
  __syncthreads();

  const int xm = tid >> 3, xc = (tid & 7) * 16;   // x-load role: 8 threads per row
  {
    // load x(t=0) into xbuf[0]; prefetch edge id for t=1
    int e = ebuf[xm];
    float xr[16];
#pragma unroll
    for (int j = 0; j < 16; ++j) xr[j] = 0.f;
    if (e >= 0) {
      const v4f* s4 = (const v4f*)(inputs + e * DIM + xc);
#pragma unroll
      for (int q = 0; q < 4; ++q) { v4f v = s4[q];
#pragma unroll
        for (int j = 0; j < 4; ++j) xr[q * 4 + j] = v[j]; }
    }
    int e1 = -1;
    if (tid < BP) e1 = (1 < lenL[tid]) ? paths[offL[tid] + 1] : -1;
    __syncthreads();
#pragma unroll
    for (int j = 0; j < 16; ++j) xL[xm * ROWP + xc + j] = (_Float16)xr[j];
    if (tid < BP) ebuf[tid] = e1;
    __syncthreads();
  }

  // per-lane constants: fused biases for this wave's columns, lens per D-row
  float bz[2], br[2], bxh[2], bhh[2];
#pragma unroll
  for (int i = 0; i < 2; ++i) {
    int col = (2 * wid + i) * 16 + ln;
    bz[i]  = gb[col] + gb[G3 + col];
    br[i]  = gb[DIM + col] + gb[G3 + DIM + col];
    bxh[i] = gb[2 * DIM + col];
    bhh[i] = gb[G3 + 2 * DIM + col];
  }
  int lenv[8];
#pragma unroll
  for (int v = 0; v < 8; ++v) lenv[v] = lenL[v + (hi << 3)];
  float hreg[2][8];
#pragma unroll
  for (int i = 0; i < 2; ++i)
#pragma unroll
    for (int v = 0; v < 8; ++v) hreg[i][v] = 0.f;

  // ---- GRU recurrence ----
#pragma clang loop unroll(disable)
  for (int t = 0; t < LMAX; ++t) {
    const int cur = t & 1, nxt = cur ^ 1;
    const bool havex = (t + 1 < LMAX);
    float xr[16];
    if (havex) {                                   // prefetch x(t+1): global loads overlap GEMMs
      int e = ebuf[xm];
#pragma unroll
      for (int j = 0; j < 16; ++j) xr[j] = 0.f;
      if (e >= 0) {
        const v4f* s4 = (const v4f*)(inputs + e * DIM + xc);
#pragma unroll
        for (int q = 0; q < 4; ++q) { v4f v = s4[q];
#pragma unroll
          for (int j = 0; j < 4; ++j) xr[q * 4 + j] = v[j]; }
      }
    }
    int e2 = -1;
    if (tid < BP && t + 2 < LMAX)
      e2 = (t + 2 < lenL[tid]) ? paths[offL[tid] + t + 2] : -1;

    // GEMM phase: zero-init accumulators (first wmma uses inline-0 SRC2);
    // biases are folded into the gate epilogue instead.
    v8f aZ[2] = {}, aR[2] = {}, aXH[2] = {}, aHH[2] = {};
    const _Float16* xcur = xL + cur * BP * ROWP;
#pragma unroll
    for (int kk = 0; kk < 4; ++kk) {
      v16h ax = load_a(xcur, lane, kk);
      v16h ah = load_a(hL, lane, kk);
#pragma unroll
      for (int i = 0; i < 2; ++i) {
        int nt = 2 * wid + i;
        aZ[i]  = wmma16(ax, load_b(wgF, kk * 24 + nt,      lane), aZ[i]);
        aZ[i]  = wmma16(ah, load_b(wrF, kk * 24 + nt,      lane), aZ[i]);
        aR[i]  = wmma16(ax, load_b(wgF, kk * 24 + 8 + nt,  lane), aR[i]);
        aR[i]  = wmma16(ah, load_b(wrF, kk * 24 + 8 + nt,  lane), aR[i]);
        aXH[i] = wmma16(ax, load_b(wgF, kk * 24 + 16 + nt, lane), aXH[i]);
        aHH[i] = wmma16(ah, load_b(wrF, kk * 24 + 16 + nt, lane), aHH[i]);
      }
    }
    // gates (D layout: VGPR v -> row m = v + 8*hi, col n = (2w+i)*16 + ln)
    float ynew[2][8];
#pragma unroll
    for (int i = 0; i < 2; ++i)
#pragma unroll
      for (int v = 0; v < 8; ++v) {
        float z = fast_sigmoid(aZ[i][v] + bz[i]);
        float r = fast_sigmoid(aR[i][v] + br[i]);
        float cand = fast_tanh(aXH[i][v] + bxh[i] + r * (aHH[i][v] + bhh[i]));
        float hn = z * hreg[i][v] + (1.f - z) * cand;
        bool act = (t < lenv[v]);
        hreg[i][v] = act ? hn : hreg[i][v];
        ynew[i][v] = act ? hn : 0.f;
      }
    __syncthreads();   // all waves done reading hL / xbuf[cur]
#pragma unroll
    for (int i = 0; i < 2; ++i) {
      int n = (2 * wid + i) * 16 + ln;
#pragma unroll
      for (int v = 0; v < 8; ++v) {
        int mm = v + (hi << 3);
        hL[mm * ROWP + n] = (_Float16)hreg[i][v];
        hsL[(t * BP + mm) * ROWP + n] = (_Float16)ynew[i][v];
      }
    }
    if (havex) {
#pragma unroll
      for (int j = 0; j < 16; ++j)
        xL[nxt * BP * ROWP + xm * ROWP + xc + j] = (_Float16)xr[j];
    }
    if (tid < BP && t + 2 < LMAX) ebuf[tid] = e2;
    __syncthreads();
  }

  // ---- attention: g = last @ Mt^T(frag), att = hs.g, s = att^T hs, out = s@Wv ----
  _Float16* mtF = (_Float16*)(smem + OFF_MT);   // overlay Wg region
  _Float16* wvF = (_Float16*)(smem + OFF_WV);
  for (int w = tid; w < 4 * 8 * 32 * 8; w += NTHREADS) {
    int word = w & 7, l2 = (w >> 3) & 31, f = w >> 8;
    int kk = f >> 3, nt = f & 7;
    int j0 = word * 2;
    int k0 = kk * 32 + ((l2 >> 4) << 4) + j0;
    int n  = nt * 16 + (l2 & 15);
    int dst = f * 512 + ((j0 >> 3) ? 256 : 0) + l2 * 8 + (j0 & 7);
    mtF[dst]     = (_Float16)mt[k0 * DIM + n];
    mtF[dst + 1] = (_Float16)mt[(k0 + 1) * DIM + n];
    wvF[dst]     = (_Float16)wvp[k0 * DIM + n];
    wvF[dst + 1] = (_Float16)wvp[(k0 + 1) * DIM + n];
  }
  __syncthreads();

  v8f aG[2] = {};
#pragma unroll
  for (int kk = 0; kk < 4; ++kk) {
    v16h ah = load_a(hL, lane, kk);
#pragma unroll
    for (int i = 0; i < 2; ++i)
      aG[i] = wmma16(ah, load_b(mtF, kk * 8 + 2 * wid + i, lane), aG[i]);
  }
#pragma unroll
  for (int i = 0; i < 2; ++i)
#pragma unroll
    for (int v = 0; v < 8; ++v)
      gL[(v + (hi << 3)) * GROWP + (2 * wid + i) * 16 + ln] = aG[i][v];
  __syncthreads();

  for (int i2 = tid; i2 < BP * LMAX; i2 += NTHREADS) {
    int p = i2 / LMAX, l = i2 - p * LMAX;
    const _Float16* hr = hsL + (l * BP + p) * ROWP;
    const float* gp = gL + p * GROWP;
    float acc = 0.f;
#pragma unroll 8
    for (int d = 0; d < DIM; ++d) acc += (float)hr[d] * gp[d];
    attL[p * LMAX + l] = acc;
  }
  __syncthreads();
  {
    int p = tid >> 3, c = (tid & 7) * 16;
    float sa[16];
#pragma unroll
    for (int j = 0; j < 16; ++j) sa[j] = 0.f;
    for (int l = 0; l < LMAX; ++l) {
      float a = attL[p * LMAX + l];
      const _Float16* hr = hsL + (l * BP + p) * ROWP + c;
#pragma unroll
      for (int j = 0; j < 16; ++j) sa[j] += a * (float)hr[j];
    }
#pragma unroll
    for (int j = 0; j < 16; ++j) sL[p * ROWP + c + j] = (_Float16)sa[j];
  }
  __syncthreads();

  v8f aC[2] = {};
#pragma unroll
  for (int kk = 0; kk < 4; ++kk) {
    v16h as = load_a(sL, lane, kk);
#pragma unroll
    for (int i = 0; i < 2; ++i)
      aC[i] = wmma16(as, load_b(wvF, kk * 8 + 2 * wid + i, lane), aC[i]);
  }
#pragma unroll
  for (int i = 0; i < 2; ++i)
#pragma unroll
    for (int v = 0; v < 8; ++v)
      out[(pbase + v + (hi << 3)) * DIM + (2 * wid + i) * 16 + ln] = aC[i][v];
}

extern "C" void kernel_launch(void* const* d_in, const int* in_sizes, int n_in,
                              void* d_out, int out_size, void* d_ws, size_t ws_size,
                              hipStream_t stream) {
  (void)in_sizes; (void)n_in; (void)out_size; (void)ws_size;
  const float* inputs = (const float*)d_in[0];
  const int*   paths  = (const int*)d_in[1];
  const int*   lens   = (const int*)d_in[4];
  const float* gk     = (const float*)d_in[5];
  const float* rk     = (const float*)d_in[6];
  const float* gb     = (const float*)d_in[7];
  const float* wq     = (const float*)d_in[8];
  const float* wk     = (const float*)d_in[9];
  const float* wvp    = (const float*)d_in[10];
  int*   offs = (int*)d_ws;                           // 64KB
  float* mtw  = (float*)((char*)d_ws + 65536);        // 64KB
  float* out  = (float*)d_out;

  hipFuncSetAttribute(reinterpret_cast<const void*>(pem_main),
                      hipFuncAttributeMaxDynamicSharedMemorySize, LDS_BYTES);
  pem_scan<<<1, 256, 0, stream>>>(lens, offs);
  pem_mt<<<64, 256, 0, stream>>>(wq, wk, mtw);
  pem_main<<<NPATH / BP, NTHREADS, LDS_BYTES, stream>>>(
      inputs, paths, lens, gk, rk, gb, wvp, offs, mtw, out);
}
